// SE3MultiSelfAttention_61821759258877
// MI455X (gfx1250) — matrix operand order repaired and checked by reading, not compiled, not run on hardware
//
#include <hip/hip_runtime.h>
#include <math.h>

typedef __attribute__((ext_vector_type(16))) _Float16 v16h;
typedef __attribute__((ext_vector_type(8)))  _Float16 v8h;
typedef __attribute__((ext_vector_type(8)))  float    v8f;

#define NB    512     // sequence length N
#define C20   20      // true head channel count
#define CP    32      // padded channel count (WMMA K)
#define BH    16      // B * NHEAD
#define NHEADS 8

struct c64 { float x, y; };

// ---------------------------------------------------------------------------
// WMMA helpers (wave32, gfx1250). A and B fragments both use contiguous k = 16*(lane/16) + s.
// ---------------------------------------------------------------------------
__device__ __forceinline__ v8f wmma_f16(v16h a, v16h b, v8f c) {
  // v_wmma_f32_16x16x32_f16 : D = A(16x32 f16) * B(32x16 f16) + C(16x16 f32)
  return __builtin_amdgcn_wmma_f32_16x16x32_f16(false, a, false, b,
                                                (short)0, c, false, false);
}

union AFrag { v16h v; struct { v8h lo, hi; } p; };

// A-matrix 16x32 f16, source S[row*stride + k] (row-major).
// lane<16: rows M=lane, K=0..15 ; lane>=16: K=16..31.
__device__ __forceinline__ v16h load_afrag(const _Float16* S, int stride, int lane) {
  int row = lane & 15;
  int k0  = (lane & 16) ? 16 : 0;
  const _Float16* p = S + (size_t)row * stride + k0;
  AFrag f;
  f.p.lo = *(const v8h*)(p);
  f.p.hi = *(const v8h*)(p + 8);
  return f.v;
}

// B-matrix 32x16 f16, logical B[k][col]; source stores S[col*stride + k].
// lane<16: K=0..15 ; lane>=16: K=16..31.
__device__ __forceinline__ v16h load_bfrag(const _Float16* S, int stride, int lane) {
  int col = lane & 15;
  int kb  = (lane & 16) ? 16 : 0;
  const _Float16* p = S + (size_t)col * stride + kb;
  return *(const v16h*)(p);
}

// ---------------------------------------------------------------------------
// K1: build Q/K/V f16 planes from emb @ x (complex GEMMs, small).
//   z=0 -> Q planes [bh][n][CP] (conj applied, scaled by 1/sqrt(160));
//          qr = Re, qi = -Im (conj), qin = +Im  (no WMMA NEG needed on f16)
//   z=1 -> K^T planes [bh][n][CP] ; z=2 -> V planes [bh][c][NB]
// c in [20,32) written as zeros (padding for WMMA K=32).
// ---------------------------------------------------------------------------
__global__ void k_build_qkv(const c64* __restrict__ x0, const c64* __restrict__ x1,
                            const c64* __restrict__ e0, const c64* __restrict__ e1,
                            _Float16* __restrict__ qr, _Float16* __restrict__ qi,
                            _Float16* __restrict__ qin,
                            _Float16* __restrict__ kr, _Float16* __restrict__ ki,
                            _Float16* __restrict__ vr, _Float16* __restrict__ vi)
{
  int blk = blockIdx.x;
  int c   = blk & 31;
  int bh  = (blk >> 5) & 15;
  int z   = blk >> 9;            // 0..2
  int b = bh >> 3, h = bh & 7;
  const float inv_scale = 0.079056941504209483f; // 1/sqrt(160)

  for (int n = threadIdx.x; n < NB; n += blockDim.x) {
    float ar = 0.f, ai = 0.f;
    if (c < C20) {
      if (c < 8) {                               // level 0: d=64, q=8, m=1
        int dd = h * 8 + c;
        const c64* em = e0 + ((size_t)z * 64 + dd) * 64;
        const c64* xp = x0 + (size_t)b * 64 * NB + n;
#pragma unroll 8
        for (int e = 0; e < 64; ++e) {
          c64 w = em[e]; c64 xv = xp[(size_t)e * NB];
          ar += w.x * xv.x - w.y * xv.y;
          ai += w.x * xv.y + w.y * xv.x;
        }
      } else {                                   // level 1: d=32, q=4, m=3
        int cl = c - 8, mi = cl >> 2, qq = cl & 3;
        int dd = h * 4 + qq;
        const c64* em = e1 + ((size_t)z * 32 + dd) * 32;
        const c64* xp = x1 + ((size_t)(b * 3 + mi) * 32) * NB + n;
#pragma unroll 8
        for (int e = 0; e < 32; ++e) {
          c64 w = em[e]; c64 xv = xp[(size_t)e * NB];
          ar += w.x * xv.x - w.y * xv.y;
          ai += w.x * xv.y + w.y * xv.x;
        }
      }
    }
    if (z == 0) {
      size_t o = ((size_t)bh * NB + n) * CP + c;
      qr[o]  = (_Float16)( ar * inv_scale);
      qi[o]  = (_Float16)(-ai * inv_scale);      // conj
      qin[o] = (_Float16)( ai * inv_scale);
    } else if (z == 1) {
      size_t o = ((size_t)bh * NB + n) * CP + c; // K^T : [m][c]
      kr[o] = (_Float16)ar;
      ki[o] = (_Float16)ai;
    } else {
      size_t o = ((size_t)bh * CP + c) * NB + n; // V : [c][j]
      vr[o] = (_Float16)ar;
      vi[o] = (_Float16)ai;
    }
  }
}

// ---------------------------------------------------------------------------
// K2: T[bh][n][g] = sum_{c in group g} conjQ[n][c] * enc[0](h, c-part)
// ---------------------------------------------------------------------------
__global__ void k_build_T(const _Float16* __restrict__ qr, const _Float16* __restrict__ qi,
                          const c64* __restrict__ enc0, const c64* __restrict__ enc1,
                          c64* __restrict__ T)
{
  int bh = blockIdx.x;
  int h  = bh & 7;
  for (int n = threadIdx.x; n < NB; n += blockDim.x) {
    const _Float16* pr = qr + ((size_t)bh * NB + n) * CP;
    const _Float16* pi = qi + ((size_t)bh * NB + n) * CP;
    float tr = 0.f, ti = 0.f;
#pragma unroll
    for (int c = 0; c < 8; ++c) {
      c64 e = enc0[(size_t)(0 * NHEADS + h) * 8 + c];
      float x = (float)pr[c], y = (float)pi[c];
      tr += x * e.x - y * e.y;
      ti += x * e.y + y * e.x;
    }
    T[((size_t)bh * NB + n) * 4 + 0] = c64{tr, ti};
#pragma unroll
    for (int g = 1; g <= 3; ++g) {
      tr = 0.f; ti = 0.f;
#pragma unroll
      for (int qq = 0; qq < 4; ++qq) {
        c64 e = enc1[(size_t)(0 * NHEADS + h) * 4 + qq];
        int c = 8 + (g - 1) * 4 + qq;
        float x = (float)pr[c], y = (float)pi[c];
        tr += x * e.x - y * e.y;
        ti += x * e.y + y * e.x;
      }
      T[((size_t)bh * NB + n) * 4 + g] = c64{tr, ti};
    }
  }
}

// ---------------------------------------------------------------------------
// K3: attention logits.  A = conjQ * K (4 WMMAs per 16x16 tile), + rank-4
// positional epilogue with constant-offset addressing, -> |A| (f32).
// __launch_bounds__(256,1): grid has 512 blocks x 8 waves, so block-level
// parallelism hides latency; let the allocator keep 3 A-frags + hoisted T
// values (~150 VGPRs) resident with zero spills.
// ---------------------------------------------------------------------------
__global__ void __launch_bounds__(256, 1)
k_attn_qk(const _Float16* __restrict__ qr, const _Float16* __restrict__ qi,
          const _Float16* __restrict__ qin,
          const _Float16* __restrict__ kr, const _Float16* __restrict__ ki,
          const c64* __restrict__ T,
          const c64* __restrict__ pos0, const c64* __restrict__ pos1,
          float* __restrict__ Alog)
{
  int lane = threadIdx.x & 31;
  int wid  = threadIdx.x >> 5;   // 0..7
  int ti   = blockIdx.x & 31;
  int bh   = blockIdx.x >> 5;

  size_t qoff = ((size_t)bh * NB + ti * 16) * CP;
  v16h aqr = load_afrag(qr  + qoff, CP, lane);
  v16h aqi = load_afrag(qi  + qoff, CP, lane);
  v16h aqn = load_afrag(qin + qoff, CP, lane);

  int rbase = ti * 16 + ((lane & 16) ? 8 : 0);   // rows handled by this lane
  const c64* t_base = T + ((size_t)bh * NB + rbase) * 4;

#pragma unroll 1
  for (int jj = 0; jj < 4; ++jj) {
    int tj = wid * 4 + jj;
    size_t koff = ((size_t)bh * NB + tj * 16) * CP;
    v16h bkr = load_bfrag(kr + koff, CP, lane);
    v16h bki = load_bfrag(ki + koff, CP, lane);

    v8f zero = {};
    // conj(Q)*K : Ar = qr*Kr + Im(Q)*Ki ; Ai = qr*Ki + (-Im(Q))*Kr
    v8f ar = wmma_f16(aqn, bki, zero);
    v8f ai = wmma_f16(aqi, bkr, zero);
    ar = wmma_f16(aqr, bkr, ar);
    ai = wmma_f16(aqr, bki, ai);

    int m = tj * 16 + (lane & 15);
    // base pointers hoisted once; all per-r accesses are constant offsets
    // (r*NB*8B <= 28KB fits the 24-bit global IOFFSET)
    const c64* p0_base = pos0 + (size_t)rbase * NB + m;
    const c64* p1_base = pos1 + (size_t)rbase * 3 * NB + m;
    float*     al_base = Alog + ((size_t)bh * NB + rbase) * NB + m;

#pragma unroll
    for (int r = 0; r < 8; ++r) {        // static acc refs, const offsets
      float xr = ar[r], xi = ai[r];
      c64 t0 = t_base[r * 4 + 0];
      c64 p0 = p0_base[(size_t)r * NB];
      xr += t0.x * p0.x - t0.y * p0.y;
      xi += t0.x * p0.y + t0.y * p0.x;
#pragma unroll
      for (int g = 1; g <= 3; ++g) {
        c64 tg = t_base[r * 4 + g];
        c64 pg = p1_base[((size_t)r * 3 + (g - 1)) * NB];
        xr += tg.x * pg.x - tg.y * pg.y;
        xi += tg.x * pg.y + tg.y * pg.x;
      }
      al_base[(size_t)r * NB] = sqrtf(xr * xr + xi * xi);
    }
  }
}

// ---------------------------------------------------------------------------
// K4: row softmax over |A|, emit f16 weights, and fold the V-side positional
// row sums:  U[bh][n][g] = sum_m P_g[n,m] * w[n,m]   (complex, 4 groups)
// grid: 16*512 blocks of 256 (one row per block)
// ---------------------------------------------------------------------------
__global__ void k_softmax(const float* __restrict__ Alog,
                          const c64* __restrict__ pos0, const c64* __restrict__ pos1,
                          _Float16* __restrict__ Asm, float* __restrict__ Uf)
{
  __shared__ float red[256];
  int row = blockIdx.x;           // bh*NB + n
  int n   = row & (NB - 1);
  int t   = threadIdx.x;
  const float* a = Alog + (size_t)row * NB;

  float v0 = a[t], v1 = a[t + 256];
  red[t] = fmaxf(v0, v1);
  __syncthreads();
  for (int s = 128; s > 0; s >>= 1) {
    if (t < s) red[t] = fmaxf(red[t], red[t + s]);
    __syncthreads();
  }
  float mx = red[0];
  __syncthreads();

  float e0 = __expf(v0 - mx), e1 = __expf(v1 - mx);
  red[t] = e0 + e1;
  __syncthreads();
  for (int s = 128; s > 0; s >>= 1) {
    if (t < s) red[t] += red[t + s];
    __syncthreads();
  }
  float inv = 1.0f / red[0];
  __syncthreads();

  float w0 = e0 * inv, w1 = e1 * inv;
  _Float16* out = Asm + (size_t)row * NB;
  out[t]       = (_Float16)w0;
  out[t + 256] = (_Float16)w1;

  float uacc[8];
  {
    c64 pa = pos0[(size_t)n * NB + t];
    c64 pb = pos0[(size_t)n * NB + t + 256];
    uacc[0] = w0 * pa.x + w1 * pb.x;
    uacc[1] = w0 * pa.y + w1 * pb.y;
#pragma unroll
    for (int mi = 0; mi < 3; ++mi) {
      c64 qa = pos1[((size_t)n * 3 + mi) * NB + t];
      c64 qb = pos1[((size_t)n * 3 + mi) * NB + t + 256];
      uacc[2 + 2 * mi] = w0 * qa.x + w1 * qb.x;
      uacc[3 + 2 * mi] = w0 * qa.y + w1 * qb.y;
    }
  }
#pragma unroll
  for (int k = 0; k < 8; ++k) {
    __syncthreads();
    red[t] = uacc[k];
    __syncthreads();
    for (int s = 128; s > 0; s >>= 1) {
      if (t < s) red[t] += red[t + s];
      __syncthreads();
    }
    if (t == 0) Uf[(size_t)row * 8 + k] = red[0];
  }
}

// ---------------------------------------------------------------------------
// K5: Vo^T[i][c] = sum_j w[i,j] * V^T[j,c].  Even/odd split accumulators ->
// 4 independent WMMA chains across the rolled k-loop, summed at the end.
// Then add enc[1](h,c) * U[bh][i][g(c)] with constant-offset addressing.
// ---------------------------------------------------------------------------
__global__ void __launch_bounds__(256, 1)
k_attn_av(const _Float16* __restrict__ Asm,
          const _Float16* __restrict__ vr, const _Float16* __restrict__ vi,
          const c64* __restrict__ U,
          const c64* __restrict__ enc0, const c64* __restrict__ enc1,
          c64* __restrict__ Vo)
{
  int lane = threadIdx.x & 31;
  int wid  = threadIdx.x >> 5;     // 0..7
  int tig  = blockIdx.x & 7;
  int bh   = blockIdx.x >> 3;
  int h    = bh & 7;
  int ti   = tig * 4 + (wid & 3);  // i-tile 0..31
  int tc   = wid >> 2;             // c-tile 0..1

  const _Float16* arow = Asm + ((size_t)bh * NB + ti * 16) * NB;
  const _Float16* vrb  = vr  + ((size_t)bh * CP + tc * 16) * NB;
  const _Float16* vib  = vi  + ((size_t)bh * CP + tc * 16) * NB;

  v8f accr0 = {}, acci0 = {}, accr1 = {}, acci1 = {};
#pragma unroll 1
  for (int kb = 0; kb < NB; kb += 64) {
    __builtin_prefetch(arow + kb + 64, 0, 0);
    v16h af0 = load_afrag(arow + kb,      NB, lane);
    v16h br0 = load_bfrag(vrb  + kb,      NB, lane);
    v16h bi0 = load_bfrag(vib  + kb,      NB, lane);
    v16h af1 = load_afrag(arow + kb + 32, NB, lane);
    v16h br1 = load_bfrag(vrb  + kb + 32, NB, lane);
    v16h bi1 = load_bfrag(vib  + kb + 32, NB, lane);
    accr0 = wmma_f16(af0, br0, accr0);
    acci0 = wmma_f16(af0, bi0, acci0);
    accr1 = wmma_f16(af1, br1, accr1);
    acci1 = wmma_f16(af1, bi1, acci1);
  }
  v8f accr = accr0 + accr1;
  v8f acci = acci0 + acci1;

  int c     = tc * 16 + (lane & 15);
  int ibase = ti * 16 + ((lane & 16) ? 8 : 0);
  bool valid = (c < C20);
  c64 e = c64{0.f, 0.f};
  int g = 0;
  if (valid) {
    if (c < 8) { e = enc0[(size_t)(1 * NHEADS + h) * 8 + c]; g = 0; }
    else {
      int cl = c - 8;
      e = enc1[(size_t)(1 * NHEADS + h) * 4 + (cl & 3)];
      g = 1 + (cl >> 2);
    }
  }
  // constant-offset bases: per-r offsets are r*4 (U) and r*CP (Vo)
  const c64* u_base  = U  + ((size_t)bh * NB + ibase) * 4 + g;
  c64*       vo_base = Vo + ((size_t)bh * NB + ibase) * CP + c;
#pragma unroll
  for (int r = 0; r < 8; ++r) {          // static acc refs, const offsets
    float xr = accr[r], xi = acci[r];
    if (valid) {
      c64 u = u_base[r * 4];
      xr += e.x * u.x - e.y * u.y;
      xi += e.x * u.y + e.y * u.x;
    }
    vo_base[r * CP] = c64{xr, xi};
  }
}

// ---------------------------------------------------------------------------
// K6: output projection.  res = w @ v per (b, mi) with the head/channel
// un-interleave folded into the gather from Vo.
// ---------------------------------------------------------------------------
__global__ void k_out_proj(const c64* __restrict__ Vo,
                           const c64* __restrict__ w0, const c64* __restrict__ w1,
                           c64* __restrict__ out)
{
  int blk = blockIdx.x;
  for (int n = threadIdx.x; n < NB; n += blockDim.x) {
    if (blk < 128) {
      int dd = blk & 63, b = blk >> 6;
      float ax = 0.f, ay = 0.f;
      const c64* wrow = w0 + (size_t)dd * 64;
#pragma unroll 8
      for (int e = 0; e < 64; ++e) {
        int h = e >> 3, qq = e & 7;              // c = qq (level-0)
        c64 w = wrow[e];
        c64 v = Vo[(((size_t)(b * 8 + h)) * NB + n) * CP + qq];
        ax += w.x * v.x - w.y * v.y;
        ay += w.x * v.y + w.y * v.x;
      }
      out[((size_t)b * 64 + dd) * NB + n] = c64{ax, ay};
    } else {
      int r2 = blk - 128;
      int dd = r2 & 31;
      int mi = (r2 >> 5) % 3;
      int b  = r2 / 96;
      float ax = 0.f, ay = 0.f;
      const c64* wrow = w1 + (size_t)dd * 32;
#pragma unroll 8
      for (int e = 0; e < 32; ++e) {
        int h = e >> 2, qq = e & 3;
        int c = 8 + mi * 4 + qq;                 // level-1 channel
        c64 w = wrow[e];
        c64 v = Vo[(((size_t)(b * 8 + h)) * NB + n) * CP + c];
        ax += w.x * v.x - w.y * v.y;
        ay += w.x * v.y + w.y * v.x;
      }
      out[(size_t)65536 + ((size_t)(b * 3 + mi) * 32 + dd) * NB + n] = c64{ax, ay};
    }
  }
}

// ---------------------------------------------------------------------------
extern "C" void kernel_launch(void* const* d_in, const int* in_sizes, int n_in,
                              void* d_out, int out_size, void* d_ws, size_t ws_size,
                              hipStream_t stream)
{
  (void)in_sizes; (void)n_in; (void)out_size; (void)ws_size;

  const c64* x0   = (const c64*)d_in[0];
  const c64* x1   = (const c64*)d_in[1];
  const c64* e0   = (const c64*)d_in[2];
  const c64* e1   = (const c64*)d_in[3];
  const c64* enc0 = (const c64*)d_in[4];
  const c64* enc1 = (const c64*)d_in[5];
  const c64* w0   = (const c64*)d_in[6];
  const c64* w1   = (const c64*)d_in[7];
  const c64* pos0 = (const c64*)d_in[8];
  const c64* pos1 = (const c64*)d_in[9];

  // workspace carve (bump allocator, 256B aligned)
  char* base = (char*)d_ws;
  size_t off = 0;
  auto alloc = [&](size_t bytes) -> void* {
    void* p = base + off;
    off += (bytes + 255) & ~(size_t)255;
    return p;
  };
  const size_t PLANE = (size_t)BH * NB * CP;         // 262144 elems
  _Float16* qr  = (_Float16*)alloc(PLANE * 2);
  _Float16* qi  = (_Float16*)alloc(PLANE * 2);
  _Float16* qin = (_Float16*)alloc(PLANE * 2);
  _Float16* kr  = (_Float16*)alloc(PLANE * 2);
  _Float16* ki  = (_Float16*)alloc(PLANE * 2);
  _Float16* vr  = (_Float16*)alloc(PLANE * 2);
  _Float16* vi  = (_Float16*)alloc(PLANE * 2);
  float*    Alog = (float*)alloc((size_t)BH * NB * NB * 4);     // 16 MB
  _Float16* Asm  = (_Float16*)alloc((size_t)BH * NB * NB * 2);  // 8 MB
  c64*      T    = (c64*)alloc((size_t)BH * NB * 4 * 8);
  c64*      U    = (c64*)alloc((size_t)BH * NB * 4 * 8);
  c64*      Vo   = (c64*)alloc((size_t)BH * NB * CP * 8);

  k_build_qkv<<<3 * BH * CP, 256, 0, stream>>>(x0, x1, e0, e1,
                                               qr, qi, qin, kr, ki, vr, vi);
  k_build_T<<<BH, 256, 0, stream>>>(qr, qi, enc0, enc1, T);
  k_attn_qk<<<BH * 32, 256, 0, stream>>>(qr, qi, qin, kr, ki, T, pos0, pos1, Alog);
  k_softmax<<<BH * NB, 256, 0, stream>>>(Alog, pos0, pos1, Asm, (float*)U);
  k_attn_av<<<BH * 8, 256, 0, stream>>>(Asm, vr, vi, U, enc0, enc1, Vo);
  k_out_proj<<<320, 256, 0, stream>>>(Vo, w0, w1, (c64*)d_out);
}
